// PerformerAttention_35923106463822
// MI455X (gfx1250) — compile-verified, matrix operand order checked by reading, not run on hardware
//
#include <hip/hip_runtime.h>

// ---------------------------------------------------------------------------
// Performer attention for MI455X (gfx1250, wave32, WMMA bf16 16x16x32).
// - All GEMMs: bf16-split-x3 WMMA (hi*hi + hi*lo + lo*hi), f32 accumulate.
// - Big GEMM tiles staged into LDS by the Tensor Data Mover (tensor_load_to_lds,
//   TENSORcnt) instead of per-thread vmem loads.
// - Causal linear attention as chunked scan: per-chunk WMMA GEMMs + tiny prefix.
// ---------------------------------------------------------------------------

typedef __attribute__((ext_vector_type(16))) __bf16   v16bf;
typedef __attribute__((ext_vector_type(8)))  float    v8f;
typedef __attribute__((ext_vector_type(4)))  unsigned u32x4;
typedef __attribute__((ext_vector_type(8)))  unsigned u32x8;

#define INV_SQRT_M 0.08838834764831845f   // 1/sqrt(128)
#define MPAD 136
#define NPAD 72

__device__ inline void bsplit(float f, __bf16& h, __bf16& l) {
    h = (__bf16)f;
    l = (__bf16)(f - (float)h);
}

// 16x16 bf16 A/B fragment load from an LDS tile laid out [row][K] row-major.
// Lane L holds row (L&15); K packing per CDNA5 ISA: elems 0..7 -> K=kb..kb+7,
// elems 8..15 -> K=kb+16..kb+23, kb = (L>>4)*8.
__device__ inline v16bf ldfrag(const __bf16* base, int stride, int lane) {
    const __bf16* p = base + (lane & 15) * stride + ((lane >> 4) << 3);
    v16bf r;
#pragma unroll
    for (int i = 0; i < 8; ++i) { r[i] = p[i]; r[i + 8] = p[i + 16]; }
    return r;
}

__device__ inline void wmma3(v8f& acc, v16bf ah, v16bf al, v16bf bh, v16bf bl) {
    acc = __builtin_amdgcn_wmma_f32_16x16x32_bf16(false, ah, false, bh, (short)0, acc, false, false);
    acc = __builtin_amdgcn_wmma_f32_16x16x32_bf16(false, ah, false, bl, (short)0, acc, false, false);
    acc = __builtin_amdgcn_wmma_f32_16x16x32_bf16(false, al, false, bh, (short)0, acc, false, false);
}

// monotone float <-> uint mapping for deterministic global max via atomicMax
__device__ inline unsigned f2key(float f) {
    unsigned b = __float_as_uint(f);
    return (b & 0x80000000u) ? ~b : (b | 0x80000000u);
}
__device__ inline float key2f(unsigned k) {
    unsigned b = (k & 0x80000000u) ? (k ^ 0x80000000u) : ~k;
    return __uint_as_float(b);
}

// ---------------------------------------------------------------------------
// Tensor Data Mover: DMA a 2D tile of f32 (tile_d0 elems x tile_d1 rows,
// row stride = stride0 elems) from global memory into LDS at lds_off.
// D# per CDNA5 ISA: group0 {count=1, lds_addr,
// global_addr[56:0], type=2}; group1 {data_size=4B, tensor dims = tile dims
// (tile guaranteed in-bounds; stride may exceed dim), tile dims, dim0 stride}.
// Tracked by TENSORcnt; EXEC is ignored, so caller gates issue per-wave.
// ---------------------------------------------------------------------------
__device__ inline void tdm_load_2d_f32(unsigned lds_off, const void* gaddr,
                                       unsigned tile_d0, unsigned tile_d1,
                                       unsigned long long stride0) {
    unsigned long long ga = (unsigned long long)gaddr;
    u32x4 g0;
    g0[0] = 1u;                                               // count=1, user D#
    g0[1] = lds_off;                                          // lds_addr (bytes)
    g0[2] = (unsigned)ga;                                     // global_addr[31:0]
    g0[3] = (unsigned)((ga >> 32) & 0x1FFFFFFu) | (2u << 30); // addr[56:32]|type=2
    u32x8 g1;
    g1[0] = 2u << 16;                                         // data_size = 4B
    g1[1] = (tile_d0 & 0xFFFFu) << 16;                        // tensor_dim0 lo
    g1[2] = (tile_d0 >> 16) | ((tile_d1 & 0xFFFFu) << 16);    // dim0 hi | dim1 lo
    g1[3] = (tile_d1 >> 16) | (tile_d0 << 16);                // dim1 hi | tile_dim0
    g1[4] = tile_d1;                                          // tile_dim1 (dim2=0)
    g1[5] = (unsigned)stride0;                                // dim0_stride[31:0]
    g1[6] = (unsigned)(stride0 >> 32) & 0xFFFFu;              // dim0_stride[47:32]
    g1[7] = 0u;                                               // dim1_stride = 0
    asm volatile("tensor_load_to_lds %0, %1" :: "s"(g0), "s"(g1) : "memory");
}

// ---------------------------------------------------------------------------
// Generic GEMM: C[M,N] = alpha*A[M,K] @ Bt[N,K]^T (+ bias[N]); M,N % 64 == 0,
// K % 32 == 0.  Block = 128 threads (4 waves), 64x64 tile, wave = 32x32.
// A/B f32 tiles staged by TDM into LDS, then split to bf16 hi/lo in LDS.
// ---------------------------------------------------------------------------
__global__ __launch_bounds__(128) void gemm_bt(const float* __restrict__ A,
                                               const float* __restrict__ Bt,
                                               const float* __restrict__ bias,
                                               float* __restrict__ C,
                                               int M, int N, int K, float alpha) {
    __shared__ float  Afs[64][32], Bfs[64][32];               // TDM landing pads
    __shared__ __bf16 Ah[64][40], Al[64][40], Bh[64][40], Bl[64][40];
    const int tid = threadIdx.x, lane = tid & 31, w = tid >> 5;
    const int tm = blockIdx.y * 64, tn = blockIdx.x * 64;
    const int wr = (w & 1) * 32, wc = (w >> 1) * 32;
    const unsigned aoff = (unsigned)(unsigned long long)(size_t)&Afs[0][0];
    const unsigned boff = (unsigned)(unsigned long long)(size_t)&Bfs[0][0];
    v8f acc[2][2] = {};
    for (int k0 = 0; k0 < K; k0 += 32) {
        __syncthreads();                    // prev split + compute done
        if (w == 0) {                       // TDM ignores EXEC: issue once (wave 0)
            tdm_load_2d_f32(aoff, A + (size_t)tm * K + k0, 32u, 64u, (unsigned)K);
            tdm_load_2d_f32(boff, Bt + (size_t)tn * K + k0, 32u, 64u, (unsigned)K);
            __builtin_amdgcn_s_wait_tensorcnt(0);
        }
        __syncthreads();                    // f32 tiles visible to all waves
#pragma unroll
        for (int i = 0; i < 16; ++i) {
            int e = tid + i * 128, r = e >> 5, kk = e & 31;
            bsplit(alpha * Afs[r][kk], Ah[r][kk], Al[r][kk]);
            bsplit(Bfs[r][kk], Bh[r][kk], Bl[r][kk]);
        }
        __syncthreads();
        v16bf ah[2], al[2], bh[2], bl[2];
#pragma unroll
        for (int i = 0; i < 2; ++i) {
            ah[i] = ldfrag(&Ah[wr + 16 * i][0], 40, lane);
            al[i] = ldfrag(&Al[wr + 16 * i][0], 40, lane);
            bh[i] = ldfrag(&Bh[wc + 16 * i][0], 40, lane);
            bl[i] = ldfrag(&Bl[wc + 16 * i][0], 40, lane);
        }
#pragma unroll
        for (int i = 0; i < 2; ++i)
#pragma unroll
            for (int j = 0; j < 2; ++j)
                wmma3(acc[i][j], ah[i], al[i], bh[j], bl[j]);
    }
#pragma unroll
    for (int i = 0; i < 2; ++i)
#pragma unroll
        for (int j = 0; j < 2; ++j) {
            int rB = tm + wr + 16 * i + ((lane >> 4) << 3);
            int cB = tn + wc + 16 * j + (lane & 15);
            float bv = bias ? bias[cB] : 0.f;
#pragma unroll
            for (int r = 0; r < 8; ++r)
                C[(size_t)(rB + r) * N + cB] = acc[i][j][r] + bv;
        }
}

// ---------------------------------------------------------------------------
// phi_q: per-row max cancels the -0.5||q||^2 term, so only proj is needed.
// One wave per row of 128.
// ---------------------------------------------------------------------------
__global__ __launch_bounds__(128) void phi_q_k(const float* __restrict__ proj,
                                               float* __restrict__ phi) {
    const int lane = threadIdx.x & 31;
    const int row = blockIdx.x * 4 + (threadIdx.x >> 5);
    const float* p = proj + (size_t)row * 128;
    float v0 = p[lane], v1 = p[lane + 32], v2 = p[lane + 64], v3 = p[lane + 96];
    float mx = fmaxf(fmaxf(v0, v1), fmaxf(v2, v3));
#pragma unroll
    for (int o = 16; o; o >>= 1) mx = fmaxf(mx, __shfl_xor(mx, o, 32));
    float* q = phi + (size_t)row * 128;
    q[lane]      = expf(v0 - mx) * INV_SQRT_M + 1e-4f;
    q[lane + 32] = expf(v1 - mx) * INV_SQRT_M + 1e-4f;
    q[lane + 64] = expf(v2 - mx) * INV_SQRT_M + 1e-4f;
    q[lane + 96] = expf(v3 - mx) * INV_SQRT_M + 1e-4f;
}

__global__ void init_gmax(unsigned* g) { *g = 0u; }

// phi_k pass1: knorm[row] = 0.5*scale^2*||k||^2 (scale^2 = 1/8) and global max
// of (rowmax(proj) - norm) via order-independent atomicMax on mapped uint.
__global__ __launch_bounds__(128) void phi_k_p1(const float* __restrict__ proj,
                                                const float* __restrict__ kflat,
                                                float* __restrict__ knorm,
                                                unsigned* __restrict__ gmax) {
    const int lane = threadIdx.x & 31;
    const int row = blockIdx.x * 4 + (threadIdx.x >> 5);
    const float* p = proj + (size_t)row * 128;
    float mx = fmaxf(fmaxf(p[lane], p[lane + 32]), fmaxf(p[lane + 64], p[lane + 96]));
    float k0 = kflat[(size_t)row * 64 + lane], k1 = kflat[(size_t)row * 64 + lane + 32];
    float s = k0 * k0 + k1 * k1;
#pragma unroll
    for (int o = 16; o; o >>= 1) {
        mx = fmaxf(mx, __shfl_xor(mx, o, 32));
        s += __shfl_xor(s, o, 32);
    }
    if (lane == 0) {
        float norm = 0.0625f * s;       // 0.5 * (1/8)
        knorm[row] = norm;
        atomicMax(gmax, f2key(mx - norm));
    }
}

__global__ __launch_bounds__(256) void phi_k_p2(const float* __restrict__ proj,
                                                const float* __restrict__ knorm,
                                                const unsigned* __restrict__ gmax,
                                                float* __restrict__ phi) {
    size_t e = (size_t)blockIdx.x * 256 + threadIdx.x;
    float g = key2f(*gmax);
    phi[e] = expf(proj[e] - knorm[e >> 7] - g) * INV_SQRT_M + 1e-4f;
}

// ---------------------------------------------------------------------------
// Stage A: per-chunk S_t[m,d] = sum_{n in chunk} phi_k[n,m]*v[n,d]  (128x64),
// plus ksum_t[m]. Block 256 = 8 waves of 32x32; K = chunk = 64.
// Grid: (t=16, bh=32).  phi layout: [((b*N+n)*H+h)*128 + m]; v: [(b*N+n)*1024+h*64+d]
// ---------------------------------------------------------------------------
__global__ __launch_bounds__(256) void stage_a(const float* __restrict__ phik,
                                               const float* __restrict__ vbuf,
                                               float* __restrict__ S,
                                               float* __restrict__ ksum) {
    const int t = blockIdx.x, bh = blockIdx.y, b = bh >> 4, h = bh & 15, n0 = t * 64;
    __shared__ __bf16 kTh[128][NPAD], kTl[128][NPAD], vTh[64][NPAD], vTl[64][NPAD];
    const int tid = threadIdx.x, lane = tid & 31, w = tid >> 5;
#pragma unroll
    for (int i = 0; i < 32; ++i) {        // phi_k chunk, transposed -> kT[m][n]
        int e = tid + i * 256, n = e >> 7, m = e & 127;
        float f = phik[((size_t)((b * 1024 + n0 + n) * 16 + h)) * 128 + m];
        bsplit(f, kTh[m][n], kTl[m][n]);
    }
#pragma unroll
    for (int i = 0; i < 16; ++i) {        // v chunk, transposed -> vT[d][n]
        int e = tid + i * 256, n = e >> 6, d = e & 63;
        float f = vbuf[(size_t)(b * 1024 + n0 + n) * 1024 + h * 64 + d];
        bsplit(f, vTh[d][n], vTl[d][n]);
    }
    __syncthreads();
    const int m0 = (w & 3) * 32, d0 = (w >> 2) * 32;
    v8f acc[2][2] = {};
#pragma unroll
    for (int n = 0; n < 64; n += 32) {
        v16bf ah[2], al[2], bvh[2], bvl[2];
#pragma unroll
        for (int i = 0; i < 2; ++i) {
            ah[i]  = ldfrag(&kTh[m0 + 16 * i][n], NPAD, lane);
            al[i]  = ldfrag(&kTl[m0 + 16 * i][n], NPAD, lane);
            bvh[i] = ldfrag(&vTh[d0 + 16 * i][n], NPAD, lane);
            bvl[i] = ldfrag(&vTl[d0 + 16 * i][n], NPAD, lane);
        }
#pragma unroll
        for (int i = 0; i < 2; ++i)
#pragma unroll
            for (int j = 0; j < 2; ++j)
                wmma3(acc[i][j], ah[i], al[i], bvh[j], bvl[j]);
    }
    float* Sp = S + ((size_t)bh * 16 + t) * 8192;
#pragma unroll
    for (int i = 0; i < 2; ++i)
#pragma unroll
        for (int j = 0; j < 2; ++j) {
            int m = m0 + 16 * i + ((lane >> 4) << 3), d = d0 + 16 * j + (lane & 15);
#pragma unroll
            for (int r = 0; r < 8; ++r) Sp[(m + r) * 64 + d] = acc[i][j][r];
        }
    if (tid < 128) {                      // exact f32 ksum (deterministic)
        float s = 0.f;
        for (int n = 0; n < 64; ++n)
            s += phik[((size_t)((b * 1024 + n0 + n) * 16 + h)) * 128 + tid];
        ksum[((size_t)bh * 16 + t) * 128 + tid] = s;
    }
}

// Stage B: exclusive prefix over the 16 chunks (per bh), in place.
__global__ __launch_bounds__(256) void stage_b(float* __restrict__ S, float* __restrict__ ksum) {
    const int bh = blockIdx.x, tid = threadIdx.x;
    for (int e = tid; e < 8192; e += 256) {
        float run = 0.f;
        for (int t = 0; t < 16; ++t) {
            float* p = S + ((size_t)bh * 16 + t) * 8192 + e;
            float tmp = *p; *p = run; run += tmp;
        }
    }
    for (int e = tid; e < 128; e += 256) {
        float run = 0.f;
        for (int t = 0; t < 16; ++t) {
            float* p = ksum + ((size_t)bh * 16 + t) * 128 + e;
            float tmp = *p; *p = run; run += tmp;
        }
    }
}

// ---------------------------------------------------------------------------
// Stage C: per chunk t:
//   P = phi_q @ phi_k^T (64x64), O1 = phi_q @ Sprev (64x64), both K=128 WMMA.
//   denom[n] = phi_q[n]·Kprev + sum_{j<=n} P[n,j];  z = 1/(denom + 1e-6)
//   Out = (O1 + tril(P) @ v) * z    (tril(P)@v via WMMA, K=64)
// Block 256 = 8 waves; wave tile 32 rows x 16 cols (2 frags).
// LDS reuse: kh/kl become masked-P, sh/sl (Sprev^T) become v^T.
// ---------------------------------------------------------------------------
__global__ __launch_bounds__(256) void stage_c(const float* __restrict__ phiq,
                                               const float* __restrict__ phik,
                                               const float* __restrict__ vbuf,
                                               const float* __restrict__ S,
                                               const float* __restrict__ ksum,
                                               float* __restrict__ attn) {
    const int t = blockIdx.x, bh = blockIdx.y, b = bh >> 4, h = bh & 15, n0 = t * 64;
    __shared__ __bf16 qh[64][MPAD], ql[64][MPAD];
    __shared__ __bf16 kh[64][MPAD], kl[64][MPAD];   // phi_k; later masked P
    __shared__ __bf16 sh[64][MPAD], sl[64][MPAD];   // Sprev^T; later v^T
    __shared__ float Pf[64][68];
    __shared__ float Kp[128];
    __shared__ float zz[64];
    const int tid = threadIdx.x, lane = tid & 31, w = tid >> 5;
    const int wr = (w & 1) * 32, c0 = (w >> 1) * 16;

#pragma unroll
    for (int i = 0; i < 32; ++i) {
        int e = tid + i * 256, n = e >> 7, m = e & 127;
        size_t gi = ((size_t)((b * 1024 + n0 + n) * 16 + h)) * 128 + m;
        bsplit(phiq[gi], qh[n][m], ql[n][m]);
        bsplit(phik[gi], kh[n][m], kl[n][m]);
    }
#pragma unroll
    for (int i = 0; i < 32; ++i) {        // Sprev^T -> sh[d][m]
        int e = tid + i * 256, m = e >> 6, d = e & 63;
        bsplit(S[((size_t)bh * 16 + t) * 8192 + m * 64 + d], sh[d][m], sl[d][m]);
    }
    if (tid < 128) Kp[tid] = ksum[((size_t)bh * 16 + t) * 128 + tid];
    __syncthreads();

    v8f accP[2] = {}, accO[2] = {};
#pragma unroll
    for (int m0 = 0; m0 < 128; m0 += 32) {
        v16bf a0h = ldfrag(&qh[wr][m0], MPAD, lane);
        v16bf a0l = ldfrag(&ql[wr][m0], MPAD, lane);
        v16bf a1h = ldfrag(&qh[wr + 16][m0], MPAD, lane);
        v16bf a1l = ldfrag(&ql[wr + 16][m0], MPAD, lane);
        v16bf bkh = ldfrag(&kh[c0][m0], MPAD, lane);
        v16bf bkl = ldfrag(&kl[c0][m0], MPAD, lane);
        v16bf bsh = ldfrag(&sh[c0][m0], MPAD, lane);
        v16bf bsl = ldfrag(&sl[c0][m0], MPAD, lane);
        wmma3(accP[0], a0h, a0l, bkh, bkl);
        wmma3(accP[1], a1h, a1l, bkh, bkl);
        wmma3(accO[0], a0h, a0l, bsh, bsl);
        wmma3(accO[1], a1h, a1l, bsh, bsl);
    }
#pragma unroll
    for (int i = 0; i < 2; ++i) {         // P frags -> LDS f32
        int rb = wr + 16 * i + ((lane >> 4) << 3), cc = c0 + (lane & 15);
#pragma unroll
        for (int r = 0; r < 8; ++r) Pf[rb + r][cc] = accP[i][r];
    }
    __syncthreads();

#pragma unroll
    for (int i = 0; i < 16; ++i) {        // masked P -> kh/kl
        int e = tid + i * 256, n = e >> 6, j = e & 63;
        float f = (j <= n) ? Pf[n][j] : 0.f;
        bsplit(f, kh[n][j], kl[n][j]);
    }
#pragma unroll
    for (int i = 0; i < 16; ++i) {        // v^T -> sh/sl
        int e = tid + i * 256, j = e >> 6, d = e & 63;
        bsplit(vbuf[(size_t)(b * 1024 + n0 + j) * 1024 + h * 64 + d], sh[d][j], sl[d][j]);
    }
    if (tid < 64) {
        int n = tid;
        float s = 0.f;
        for (int j = 0; j <= n; ++j) s += Pf[n][j];
        float s2 = 0.f;
        for (int m = 0; m < 128; ++m) s2 += ((float)qh[n][m] + (float)ql[n][m]) * Kp[m];
        zz[n] = 1.f / (s + s2 + 1e-6f);
    }
    __syncthreads();

#pragma unroll
    for (int j0 = 0; j0 < 64; j0 += 32) { // tril(P) @ v, accumulate into accO
        v16bf a0h = ldfrag(&kh[wr][j0], MPAD, lane);
        v16bf a0l = ldfrag(&kl[wr][j0], MPAD, lane);
        v16bf a1h = ldfrag(&kh[wr + 16][j0], MPAD, lane);
        v16bf a1l = ldfrag(&kl[wr + 16][j0], MPAD, lane);
        v16bf bvh = ldfrag(&sh[c0][j0], MPAD, lane);
        v16bf bvl = ldfrag(&sl[c0][j0], MPAD, lane);
        wmma3(accO[0], a0h, a0l, bvh, bvl);
        wmma3(accO[1], a1h, a1l, bvh, bvl);
    }
#pragma unroll
    for (int i = 0; i < 2; ++i) {
        int rb = wr + 16 * i + ((lane >> 4) << 3), cc = c0 + (lane & 15);
#pragma unroll
        for (int r = 0; r < 8; ++r) {
            int n = rb + r;
            attn[(size_t)(b * 1024 + n0 + n) * 1024 + h * 64 + cc] = accO[i][r] * zz[n];
        }
    }
}

// ---------------------------------------------------------------------------
extern "C" void kernel_launch(void* const* d_in, const int* in_sizes, int n_in,
                              void* d_out, int out_size, void* d_ws, size_t ws_size,
                              hipStream_t stream) {
    const float* x     = (const float*)d_in[0];
    const float* omega = (const float*)d_in[1];
    const float* Wq    = (const float*)d_in[2];
    const float* Wk    = (const float*)d_in[3];
    const float* Wv    = (const float*)d_in[4];
    const float* Wo    = (const float*)d_in[5];
    const float* bo    = (const float*)d_in[6];
    float* out = (float*)d_out;

    // workspace layout (floats)
    float* qbuf  = (float*)d_ws;                    // [2048,1024] (reused as attn)
    float* kbuf  = qbuf  + 2048 * 1024;             // [2048,1024]
    float* vbuf  = kbuf  + 2048 * 1024;             // [2048,1024]
    float* qproj = vbuf  + 2048 * 1024;             // [32768,128]
    float* kproj = qproj + 32768 * 128;             // [32768,128]
    float* phiq  = kproj + 32768 * 128;             // [32768,128]
    float* phik  = phiq  + 32768 * 128;             // [32768,128]
    float* S     = phik  + 32768 * 128;             // [32][16][128][64]
    float* ksumb = S     + 32 * 16 * 8192;          // [32][16][128]
    float* knorm = ksumb + 32 * 16 * 128;           // [32768]
    unsigned* gmax = (unsigned*)(knorm + 32768);    // [1]
    float* attn = qbuf;                             // alias (qbuf dead by then)

    const float SCALE = 0.35355339059327373f;       // 64^-0.25

    dim3 gBig(16, 32);   // N/64=16, M/64=32 for 2048x1024 outputs
    dim3 gPrj(2, 512);   // 128/64, 32768/64
    dim3 gChk(16, 32);   // (t, bh)

    // QKV projections
    gemm_bt<<<gBig, 128, 0, stream>>>(x, Wq, nullptr, qbuf, 2048, 1024, 1024, 1.f);
    gemm_bt<<<gBig, 128, 0, stream>>>(x, Wk, nullptr, kbuf, 2048, 1024, 1024, 1.f);
    gemm_bt<<<gBig, 128, 0, stream>>>(x, Wv, nullptr, vbuf, 2048, 1024, 1024, 1.f);
    // feature projections (scale folded into alpha): [32768,64] @ omega[128,64]^T
    gemm_bt<<<gPrj, 128, 0, stream>>>(qbuf, omega, nullptr, qproj, 32768, 128, 64, SCALE);
    gemm_bt<<<gPrj, 128, 0, stream>>>(kbuf, omega, nullptr, kproj, 32768, 128, 64, SCALE);
    // phi
    phi_q_k<<<8192, 128, 0, stream>>>(qproj, phiq);
    init_gmax<<<1, 1, 0, stream>>>(gmax);
    phi_k_p1<<<8192, 128, 0, stream>>>(kproj, kbuf, knorm, gmax);
    phi_k_p2<<<16384, 256, 0, stream>>>(kproj, knorm, gmax, phik);
    // chunked causal linear attention
    stage_a<<<gChk, 256, 0, stream>>>(phik, vbuf, S, ksumb);
    stage_b<<<32, 256, 0, stream>>>(S, ksumb);
    stage_c<<<gChk, 256, 0, stream>>>(phiq, phik, vbuf, S, ksumb, attn);
    // output projection + bias
    gemm_bt<<<gBig, 128, 0, stream>>>(attn, Wo, bo, out, 2048, 1024, 1024, 1.f);
}
